// DeepClusteringLoss_17386027614820
// MI455X (gfx1250) — matrix operand order repaired and checked by reading, not compiled, not run on hardware
//
#include <hip/hip_runtime.h>
#include <hip/hip_bf16.h>

typedef __attribute__((ext_vector_type(16))) __bf16   v16bf;
typedef __attribute__((ext_vector_type(8)))  float    v8f;
typedef __attribute__((ext_vector_type(4)))  unsigned int u32x4;

namespace {
constexpr int kB      = 16;
constexpr int kT      = 16384;
constexpr int kD      = 256;
constexpr int kS      = 4;
constexpr int kChunks = 8;
constexpr int kTC     = kT / kChunks;   // 2048 time rows per chunk
constexpr int kKT     = 32;             // rows per LDS tile = one WMMA K-step (bf16 K=32)
constexpr int kNT     = kTC / kKT;      // 64 tiles per chunk
constexpr int kPad    = 40;             // bf16 elems per padded LDS row (80 B -> conflict-free b128)
}

union FragU {
  unsigned short s[16];
  u32x4 u[2];
  v16bf v;
};

// A (16x32 bf16): lane<16 holds M=lane, K=0..15 -> bytes 0..31
//                 lane>=16 holds M=lane-16, K=16..31 -> bytes 32..63
__device__ __forceinline__ void loadA(FragU& f, const __hip_bfloat16 (*vt)[kPad],
                                      int i, int c, int hb) {
  const char* ab = (const char*)&vt[i * 16 + c][0] + hb * 32;
  f.u[0] = *(const u32x4*)ab;
  f.u[1] = *(const u32x4*)(ab + 16);
}

// B (32x16 bf16): lane<16 holds N=lane, K=0..15 -> bytes 0..31;
//                 lane>=16 holds N=lane-16, K=16..31 -> bytes 32..63.
__device__ __forceinline__ void loadB(FragU& f, const __hip_bfloat16 (*vt)[kPad],
                                      int j, int c, int hb) {
  const char* bb = (const char*)&vt[j * 16 + c][0] + hb * 32;
  f.u[0] = *(const u32x4*)bb;
  f.u[1] = *(const u32x4*)(bb + 16);
}

// ---------------------------------------------------------------------------
// Kernel 1: per (chunk, batch) partial Grams  Gvv = V^T V  and  Gvy = V^T Y.
// 512 threads = 16 waves in a 4x4 grid of patches; wave w computes the 4x4
// block patch rows [4*(w>>2) .. +3], cols [4*(w&3) .. +3]: 4 A-frags + 4
// B-frags -> 16 WMMAs (~1 ds_load_b128 per WMMA), plus its Gvy block.
// ---------------------------------------------------------------------------
__global__ __launch_bounds__(512, 1)
void dcl_gram(const float* __restrict__ emb,
              const float* __restrict__ labels,
              float* __restrict__ gvvP,
              float* __restrict__ gvyP)
{
  __shared__ __align__(16) __hip_bfloat16 vt[kD][kPad];  // transposed tile: vt[d][t]
  __shared__ int lab[kKT];

  const int chunk = blockIdx.x;
  const int b     = blockIdx.y;
  const int tid   = threadIdx.x;
  const int lane  = tid & 31;
  const int w     = tid >> 5;     // wave id 0..15
  const int hb    = lane >> 4;    // half-wave (0/1)
  const int c     = lane & 15;    // 0..15 (col group / N index / M index)
  const int r     = 2 * w + hb;   // time row handled during the load phase
  const int wr    = (w >> 2) * 4; // base row-block of this wave's 4x4 patch
  const int wc    = (w & 3) * 4;  // base col-block of this wave's 4x4 patch

  v8f acc[17];   // 16 Gvv patch blocks (ri*4+ci) + 1 Gvy block
#pragma unroll
  for (int i = 0; i < 17; ++i) acc[i] = (v8f){0,0,0,0,0,0,0,0};

  const float* ebase = emb    + ((size_t)b * kT + (size_t)chunk * kTC) * kD;
  const float* lbase = labels + ((size_t)b * kT + (size_t)chunk * kTC) * kS;

  // ---- prologue: load tile 0 ----
  float4 x0, x1, x2, x3;
  {
    const float4* e4 = (const float4*)(ebase + (size_t)r * kD + c * 16);
    x0 = e4[0]; x1 = e4[1]; x2 = e4[2]; x3 = e4[3];
  }
  int cls = 0;
  if (c == 0) {
    const float4 l = *(const float4*)(lbase + (size_t)r * kS);
    cls = (int)(l.y + 2.0f * l.z + 3.0f * l.w + 0.5f);
  }

#pragma unroll 1
  for (int tile = 0; tile < kNT; ++tile) {
    // ---- row-normalize the resident 32x256 f32 tile ----
    float ss = x0.x*x0.x + x0.y*x0.y + x0.z*x0.z + x0.w*x0.w
             + x1.x*x1.x + x1.y*x1.y + x1.z*x1.z + x1.w*x1.w
             + x2.x*x2.x + x2.y*x2.y + x2.z*x2.z + x2.w*x2.w
             + x3.x*x3.x + x3.y*x3.y + x3.z*x3.z + x3.w*x3.w;
    ss += __shfl_xor(ss, 1);
    ss += __shfl_xor(ss, 2);
    ss += __shfl_xor(ss, 4);
    ss += __shfl_xor(ss, 8);
    const float inv = 1.0f / fmaxf(sqrtf(ss), 1e-12f);  // v / max(||v||, eps)

    __syncthreads();  // previous tile's WMMA reads are done

    {
      const float vv[16] = { x0.x, x0.y, x0.z, x0.w, x1.x, x1.y, x1.z, x1.w,
                             x2.x, x2.y, x2.z, x2.w, x3.x, x3.y, x3.z, x3.w };
#pragma unroll
      for (int j = 0; j < 16; ++j)
        vt[c * 16 + j][r] = __float2bfloat16(vv[j] * inv);
      if (c == 0) lab[r] = cls;
    }

    __syncthreads();  // transposed bf16 tile visible to all waves

    // ---- prefetch next tile's globals; latency hidden under WMMA phase ----
    if (tile + 1 < kNT) {
      const float4* n4 =
          (const float4*)(ebase + (size_t)((tile + 1) * kKT + r) * kD + c * 16);
      x0 = n4[0]; x1 = n4[1]; x2 = n4[2]; x3 = n4[3];
      if (c == 0) {
        const float4 l =
            *(const float4*)(lbase + (size_t)((tile + 1) * kKT + r) * kS);
        cls = (int)(l.y + 2.0f * l.z + 3.0f * l.w + 0.5f);
      }
    }

    // ---- WMMA phase: 4x4 register-tiled patch ----
    FragU fa[4];
#pragma unroll
    for (int ri = 0; ri < 4; ++ri) loadA(fa[ri], vt, wr + ri, c, hb);

#pragma unroll
    for (int ci = 0; ci < 4; ++ci) {
      FragU fb;
      loadB(fb, vt, wc + ci, c, hb);
#pragma unroll
      for (int ri = 0; ri < 4; ++ri)
        acc[ri * 4 + ci] = __builtin_amdgcn_wmma_f32_16x16x32_bf16(
            false, fa[ri].v, false, fb.v, (short)0, acc[ri * 4 + ci], false, false);
    }

    // ---- Gvy block for row-block w: A row w  x  one-hot Y (exact in bf16) ----
    {
      FragU faw, fy;
      loadA(faw, vt, w, c, hb);
      const int ks = hb * 16;
#pragma unroll
      for (int k = 0; k < 16; ++k)
        fy.s[k] = (lab[ks + k] == c) ? (unsigned short)0x3F80 : (unsigned short)0;
      acc[16] = __builtin_amdgcn_wmma_f32_16x16x32_bf16(
          false, faw.v, false, fy.v, (short)0, acc[16], false, false);
    }
  }

  // ---- write this chunk's partial Grams (plain stores, no atomics) ----
  // C/D layout: lane<16 -> N=lane, M=rr ; lane>=16 -> N=lane-16, M=rr+8
  float* gvv = gvvP + (size_t)(chunk * kB + b) * kD * kD;
  float* gvy = gvyP + (size_t)(chunk * kB + b) * kD * 16;
#pragma unroll
  for (int rr = 0; rr < 8; ++rr) {
    const int m = rr + hb * 8;
#pragma unroll
    for (int ri = 0; ri < 4; ++ri)
#pragma unroll
      for (int ci = 0; ci < 4; ++ci)
        gvv[(size_t)((wr + ri) * 16 + m) * kD + (wc + ci) * 16 + c] =
            acc[ri * 4 + ci][rr];
    gvy[(w * 16 + m) * 16 + c] = acc[16][rr];
  }
}

// ---------------------------------------------------------------------------
// Kernel 2: per-batch reduction: sum partial Grams, square-Frobenius, counts
// ---------------------------------------------------------------------------
__global__ __launch_bounds__(256)
void dcl_finalize(const float* __restrict__ gvvP,
                  const float* __restrict__ gvyP,
                  const float* __restrict__ labels,
                  float* __restrict__ perb)
{
  const int b   = blockIdx.x;
  const int tid = threadIdx.x;  // 256 threads
  __shared__ float red[6][256];

  float a1 = 0.f;  // ||Gvv||_F^2
  for (int idx = tid; idx < kD * kD; idx += 256) {
    float g = 0.f;
#pragma unroll
    for (int ch = 0; ch < kChunks; ++ch)
      g += gvvP[(size_t)(ch * kB + b) * kD * kD + idx];
    a1 += g * g;
  }
  float a2 = 0.f;  // ||Gvy||_F^2 (padded cols 4..15 are exactly zero)
  for (int idx = tid; idx < kD * 16; idx += 256) {
    float g = 0.f;
#pragma unroll
    for (int ch = 0; ch < kChunks; ++ch)
      g += gvyP[(size_t)(ch * kB + b) * kD * 16 + idx];
    a2 += g * g;
  }
  float c0 = 0.f, c1 = 0.f, c2 = 0.f, c3 = 0.f;  // class counts (Y^T Y is diag)
  for (int t = tid; t < kT; t += 256) {
    const float4 l = *(const float4*)(labels + ((size_t)b * kT + t) * kS);
    c0 += l.x; c1 += l.y; c2 += l.z; c3 += l.w;
  }

  red[0][tid] = a1; red[1][tid] = a2;
  red[2][tid] = c0; red[3][tid] = c1; red[4][tid] = c2; red[5][tid] = c3;
  __syncthreads();
  for (int s = 128; s > 0; s >>= 1) {
    if (tid < s) {
#pragma unroll
      for (int k = 0; k < 6; ++k) red[k][tid] += red[k][tid + s];
    }
    __syncthreads();
  }
  if (tid == 0) {
    const float gyy2 = red[2][0]*red[2][0] + red[3][0]*red[3][0]
                     + red[4][0]*red[4][0] + red[5][0]*red[5][0];
    perb[b] = (red[0][0] - 2.0f * red[1][0] + gyy2) / ((float)kT * (float)kT);
  }
}

// ---------------------------------------------------------------------------
// Kernel 3: mean over batches -> scalar output
// ---------------------------------------------------------------------------
__global__ void dcl_mean(const float* __restrict__ perb, float* __restrict__ out)
{
  if (threadIdx.x == 0 && blockIdx.x == 0) {
    float s = 0.f;
    for (int i = 0; i < kB; ++i) s += perb[i];
    out[0] = s / (float)kB;
  }
}

extern "C" void kernel_launch(void* const* d_in, const int* in_sizes, int n_in,
                              void* d_out, int out_size, void* d_ws, size_t ws_size,
                              hipStream_t stream)
{
  (void)in_sizes; (void)n_in; (void)out_size; (void)ws_size;
  const float* emb    = (const float*)d_in[0];  // [B,T,D] f32
  const float* labels = (const float*)d_in[1];  // [B,T,S] f32 one-hot
  float* out = (float*)d_out;

  // workspace layout (floats): Gvv partials | Gvy partials | per-batch loss
  float* gvvP = (float*)d_ws;                                    // 8*16*256*256
  float* gvyP = gvvP + (size_t)kChunks * kB * kD * kD;           // 8*16*256*16
  float* perb = gvyP + (size_t)kChunks * kB * kD * 16;           // 16
  // total ~35.7 MB

  dim3 grid(kChunks, kB);
  dcl_gram<<<grid, 512, 0, stream>>>(emb, labels, gvvP, gvyP);
  dcl_finalize<<<kB, 256, 0, stream>>>(gvvP, gvyP, labels, perb);
  dcl_mean<<<1, 32, 0, stream>>>(perb, out);
}
